// Attention_regular_54065048322569
// MI455X (gfx1250) — matrix operand order repaired and checked by reading, not compiled, not run on hardware
//
#include <hip/hip_runtime.h>

typedef __attribute__((ext_vector_type(16))) _Float16 v16h;
typedef __attribute__((ext_vector_type(8)))  float    v8f;
typedef __attribute__((ext_vector_type(4)))  float    v4f;   // native vector for NT loads

#define NUM_HEADS 6
#define POSD 12
#define NBIAS 945
#define NTOK 256          // tokens per window (8x32)
#define HD 32             // head dim
#define LTOT 65536        // H*W
#define CTOT 192
#define VPAD 264          // padded row length for transposed V (bank stagger)

// ---------------------------------------------------------------------------
// Kernel 1: DynamicPosBias MLP.  945 rows, trivial VALU work.
// ---------------------------------------------------------------------------
__device__ __forceinline__ void ln_relu12(const float* x, const float* g,
                                          const float* b, float* y) {
    float m = 0.f;
#pragma unroll
    for (int i = 0; i < POSD; ++i) m += x[i];
    m *= (1.0f / POSD);
    float v = 0.f;
#pragma unroll
    for (int i = 0; i < POSD; ++i) { float d = x[i] - m; v += d * d; }
    v *= (1.0f / POSD);
    float r = rsqrtf(v + 1e-5f);
#pragma unroll
    for (int i = 0; i < POSD; ++i) {
        float t = (x[i] - m) * r * g[i] + b[i];
        y[i] = t > 0.f ? t : 0.f;
    }
}

__global__ void dyn_pos_kernel(const float* __restrict__ rpe,
                               const float* __restrict__ ppw, const float* __restrict__ ppb,
                               const float* __restrict__ g1, const float* __restrict__ b1,
                               const float* __restrict__ w1, const float* __restrict__ bb1,
                               const float* __restrict__ g2, const float* __restrict__ b2,
                               const float* __restrict__ w2, const float* __restrict__ bb2,
                               const float* __restrict__ g3, const float* __restrict__ b3,
                               const float* __restrict__ w3, const float* __restrict__ bb3,
                               float* __restrict__ pos) {
    int r = blockIdx.x * blockDim.x + threadIdx.x;
    if (r >= NBIAS) return;
    float x0 = rpe[2 * r], x1 = rpe[2 * r + 1];
    float a[POSD], y[POSD];
#pragma unroll
    for (int i = 0; i < POSD; ++i)
        a[i] = ppw[2 * i] * x0 + ppw[2 * i + 1] * x1 + ppb[i];
    ln_relu12(a, g1, b1, y);
#pragma unroll
    for (int i = 0; i < POSD; ++i) {
        float s = bb1[i];
#pragma unroll
        for (int j = 0; j < POSD; ++j) s += w1[i * POSD + j] * y[j];
        a[i] = s;
    }
    ln_relu12(a, g2, b2, y);
#pragma unroll
    for (int i = 0; i < POSD; ++i) {
        float s = bb2[i];
#pragma unroll
        for (int j = 0; j < POSD; ++j) s += w2[i * POSD + j] * y[j];
        a[i] = s;
    }
    ln_relu12(a, g3, b3, y);
#pragma unroll
    for (int h = 0; h < NUM_HEADS; ++h) {
        float s = bb3[h];
#pragma unroll
        for (int j = 0; j < POSD; ++j) s += w3[h * POSD + j] * y[j];
        pos[r * NUM_HEADS + h] = s;
    }
}

// ---------------------------------------------------------------------------
// Kernel 2: expand rel_bias[h][q][k] = pos[rpi[q,k]][h]   (6 x 256 x 256 f32)
// ---------------------------------------------------------------------------
__global__ void bias_expand_kernel(const int* __restrict__ rpi,
                                   const float* __restrict__ pos,
                                   float* __restrict__ rel) {
    int t = blockIdx.x * blockDim.x + threadIdx.x;   // 0 .. 65535
    int idx = rpi[t];
#pragma unroll
    for (int h = 0; h < NUM_HEADS; ++h)
        rel[h * LTOT + t] = pos[idx * NUM_HEADS + h];
}

// ---------------------------------------------------------------------------
// Kernel 3: two-pass (exact-max) attention per (window, head).
// Block = 256 threads = 8 wave32; each wave owns 32 query rows (2 M-tiles).
// Pass 1: S = QK^T + bias via WMMA, per-lane row-max (no shuffles in loop).
// Pass 2: recompute S, P = exp(S - rowmax), per-lane row-sum, O += P@V (WMMA).
// Cross-lane butterflies only at pass boundaries: 128 bpermutes/wave total.
// ---------------------------------------------------------------------------
__global__ __launch_bounds__(256) void attn_kernel(const float* __restrict__ qkv,
                                                   const float* __restrict__ rel,
                                                   float* __restrict__ out) {
    __shared__ _Float16 Qs[NTOK][HD];
    __shared__ _Float16 Ks[NTOK][HD];
    __shared__ _Float16 VsT[HD][VPAD];    // V transposed: [dim][key], padded rows
    __shared__ _Float16 Pst[8][16][48];   // per-wave P staging (padded rows)

    const int wh = blockIdx.x;
    const int w  = wh / NUM_HEADS;
    const int h  = wh % NUM_HEADS;
    const int b  = w >> 8;            // window = b*256 + hi*8 + wi
    const int rem = w & 255;
    const int hwin = rem >> 3;        // hi in [0,32)
    const int wwin = rem & 7;         // wi in [0,8)

    const int t = threadIdx.x;

    // ---- cooperative load of Q/K/V window into LDS as f16 (scale folded into Q)
    {
        const int n    = t;                        // token in window
        const int grow = hwin * 8  + (n >> 5);
        const int gcol = wwin * 32 + (n & 31);
        const size_t pix = (size_t)grow * 256 + gcol;
        const float scale = 0.17677669529663687f;  // 1/sqrt(32)
#pragma unroll
        for (int m = 0; m < 3; ++m) {
            const size_t base = (((size_t)m * 2 + b) * LTOT + pix) * CTOT + h * HD;
            const v4f* src = (const v4f*)(qkv + base);   // 128B aligned
#pragma unroll
            for (int i = 0; i < 8; ++i) {
                v4f f = __builtin_nontemporal_load(src + i);  // read-once stream
                if (m == 0) {
                    Qs[n][i * 4 + 0] = (_Float16)(f[0] * scale);
                    Qs[n][i * 4 + 1] = (_Float16)(f[1] * scale);
                    Qs[n][i * 4 + 2] = (_Float16)(f[2] * scale);
                    Qs[n][i * 4 + 3] = (_Float16)(f[3] * scale);
                } else if (m == 1) {
                    Ks[n][i * 4 + 0] = (_Float16)f[0];
                    Ks[n][i * 4 + 1] = (_Float16)f[1];
                    Ks[n][i * 4 + 2] = (_Float16)f[2];
                    Ks[n][i * 4 + 3] = (_Float16)f[3];
                } else {
                    VsT[i * 4 + 0][n] = (_Float16)f[0];
                    VsT[i * 4 + 1][n] = (_Float16)f[1];
                    VsT[i * 4 + 2][n] = (_Float16)f[2];
                    VsT[i * 4 + 3][n] = (_Float16)f[3];
                }
            }
        }
    }
    __syncthreads();

    const int wave = t >> 5;
    const int lane = t & 31;
    const int l16  = lane & 15;
    const int hhf  = lane >> 4;       // 0 = lanes 0-15, 1 = lanes 16-31

    // ---- Q A-operands (16x32 f16, contiguous k = 16*hhf + i) for both M-tiles
    v16h qa[2];
#pragma unroll
    for (int mt = 0; mt < 2; ++mt) {
        const int mrow = (wave * 2 + mt) * 16 + l16;
        const _Float16* s0 = &Qs[mrow][16 * hhf];
        const _Float16* s1 = &Qs[mrow][16 * hhf + 8];
#pragma unroll
        for (int i = 0; i < 8; ++i) { qa[mt][i] = s0[i]; qa[mt][8 + i] = s1[i]; }
    }

    const float* biasH = rel + (size_t)h * LTOT;

    // =========================== PASS 1: row max ===========================
    float rmax[2][8];
#pragma unroll
    for (int mt = 0; mt < 2; ++mt)
#pragma unroll
        for (int j = 0; j < 8; ++j) rmax[mt][j] = -1e30f;

    for (int kt = 0; kt < 8; ++kt) {
        const int kb = kt * 32;
        v16h kbt0, kbt1;
        {
            const _Float16* s0 = &Ks[kb + l16][16 * hhf];
            const _Float16* s1 = &Ks[kb + 16 + l16][16 * hhf];
#pragma unroll
            for (int i = 0; i < 16; ++i) { kbt0[i] = s0[i]; kbt1[i] = s1[i]; }
        }
#pragma unroll
        for (int mt = 0; mt < 2; ++mt) {
            const int mbase = (wave * 2 + mt) * 16;
            v8f c0, c1;
#pragma unroll
            for (int j = 0; j < 8; ++j) {
                const int qrow = mbase + j + 8 * hhf;
                c0[j] = biasH[(size_t)qrow * 256 + kb + l16];
                c1[j] = biasH[(size_t)qrow * 256 + kb + 16 + l16];
            }
            c0 = __builtin_amdgcn_wmma_f32_16x16x32_f16(false, qa[mt], false, kbt0,
                                                        (short)0, c0, false, false);
            c1 = __builtin_amdgcn_wmma_f32_16x16x32_f16(false, qa[mt], false, kbt1,
                                                        (short)0, c1, false, false);
#pragma unroll
            for (int j = 0; j < 8; ++j)
                rmax[mt][j] = fmaxf(rmax[mt][j], fmaxf(c0[j], c1[j]));
        }
    }
    // single cross-lane butterfly per row (within 16-lane halves)
#pragma unroll
    for (int mt = 0; mt < 2; ++mt)
#pragma unroll
        for (int j = 0; j < 8; ++j) {
            float mx = rmax[mt][j];
#pragma unroll
            for (int m = 8; m >= 1; m >>= 1) mx = fmaxf(mx, __shfl_xor(mx, m, 32));
            rmax[mt][j] = mx;
        }

    // ====================== PASS 2: exp / sum / P@V ========================
    v8f o[2][2] = {};                 // [M-tile][dim-half] f32 accumulators
    float lsum[2][8];
#pragma unroll
    for (int mt = 0; mt < 2; ++mt)
#pragma unroll
        for (int j = 0; j < 8; ++j) lsum[mt][j] = 0.f;

    for (int kt = 0; kt < 8; ++kt) {
        const int kb = kt * 32;
        v16h kbt0, kbt1;
        {
            const _Float16* s0 = &Ks[kb + l16][16 * hhf];
            const _Float16* s1 = &Ks[kb + 16 + l16][16 * hhf];
#pragma unroll
            for (int i = 0; i < 16; ++i) { kbt0[i] = s0[i]; kbt1[i] = s1[i]; }
        }
        // V B-operands from transposed LDS: contiguous 32B per lane
        v16h vb0, vb1;
        {
            const _Float16* sv0 = &VsT[l16][kb + 16 * hhf];
            const _Float16* sv1 = &VsT[16 + l16][kb + 16 * hhf];
#pragma unroll
            for (int i = 0; i < 16; ++i) { vb0[i] = sv0[i]; vb1[i] = sv1[i]; }
        }
#pragma unroll
        for (int mt = 0; mt < 2; ++mt) {
            const int mbase = (wave * 2 + mt) * 16;
            v8f c0, c1;
#pragma unroll
            for (int j = 0; j < 8; ++j) {
                const int qrow = mbase + j + 8 * hhf;
                c0[j] = biasH[(size_t)qrow * 256 + kb + l16];
                c1[j] = biasH[(size_t)qrow * 256 + kb + 16 + l16];
            }
            c0 = __builtin_amdgcn_wmma_f32_16x16x32_f16(false, qa[mt], false, kbt0,
                                                        (short)0, c0, false, false);
            c1 = __builtin_amdgcn_wmma_f32_16x16x32_f16(false, qa[mt], false, kbt1,
                                                        (short)0, c1, false, false);
            // exp with exact row max; per-lane partial row sums (no shuffles)
#pragma unroll
            for (int j = 0; j < 8; ++j) {
                const float p0 = __expf(c0[j] - rmax[mt][j]);
                const float p1 = __expf(c1[j] - rmax[mt][j]);
                lsum[mt][j] += p0 + p1;
                c0[j] = p0; c1[j] = p1;
            }
            // C-layout -> A-layout conversion for P via per-wave LDS staging
#pragma unroll
            for (int j = 0; j < 8; ++j) {
                const int pr = j + 8 * hhf;
                Pst[wave][pr][l16]      = (_Float16)c0[j];
                Pst[wave][pr][16 + l16] = (_Float16)c1[j];
            }
            v16h pa;
            {
                const _Float16* s0 = &Pst[wave][l16][16 * hhf];
                const _Float16* s1 = &Pst[wave][l16][16 * hhf + 8];
#pragma unroll
                for (int i = 0; i < 8; ++i) { pa[i] = s0[i]; pa[8 + i] = s1[i]; }
            }
            o[mt][0] = __builtin_amdgcn_wmma_f32_16x16x32_f16(false, pa, false, vb0,
                                                              (short)0, o[mt][0], false, false);
            o[mt][1] = __builtin_amdgcn_wmma_f32_16x16x32_f16(false, pa, false, vb1,
                                                              (short)0, o[mt][1], false, false);
        }
    }

    // ---- epilogue: reduce row sums, normalize, NT-scatter to (B,H,W,C)
#pragma unroll
    for (int mt = 0; mt < 2; ++mt) {
#pragma unroll
        for (int j = 0; j < 8; ++j) {
            float rs = lsum[mt][j];
#pragma unroll
            for (int m = 8; m >= 1; m >>= 1) rs += __shfl_xor(rs, m, 32);
            const float inv = 1.0f / rs;
            const int qrow = (wave * 2 + mt) * 16 + j + 8 * hhf;   // token in window
            const int grow = hwin * 8  + (qrow >> 5);
            const int gcol = wwin * 32 + (qrow & 31);
            const size_t obase = (((size_t)b * 256 + grow) * 256 + gcol) * CTOT + h * HD;
            __builtin_nontemporal_store(o[mt][0][j] * inv, out + obase + l16);
            __builtin_nontemporal_store(o[mt][1][j] * inv, out + obase + 16 + l16);
        }
    }
}

// ---------------------------------------------------------------------------
extern "C" void kernel_launch(void* const* d_in, const int* in_sizes, int n_in,
                              void* d_out, int out_size, void* d_ws, size_t ws_size,
                              hipStream_t stream) {
    const float* qkv = (const float*)d_in[0];
    const int*   rpi = (const int*)d_in[3];
    const float* rpe = (const float*)d_in[4];
    const float* ppw = (const float*)d_in[5];
    const float* ppb = (const float*)d_in[6];
    const float* g1  = (const float*)d_in[7];
    const float* b1  = (const float*)d_in[8];
    const float* w1  = (const float*)d_in[9];
    const float* bb1 = (const float*)d_in[10];
    const float* g2  = (const float*)d_in[11];
    const float* b2  = (const float*)d_in[12];
    const float* w2  = (const float*)d_in[13];
    const float* bb2 = (const float*)d_in[14];
    const float* g3  = (const float*)d_in[15];
    const float* b3  = (const float*)d_in[16];
    const float* w3  = (const float*)d_in[17];
    const float* bb3 = (const float*)d_in[18];

    float* rel = (float*)d_ws;                                    // 6*65536 f32
    float* pos = (float*)((char*)d_ws + (size_t)NUM_HEADS * LTOT * sizeof(float));

    dyn_pos_kernel<<<(NBIAS + 255) / 256, 256, 0, stream>>>(
        rpe, ppw, ppb, g1, b1, w1, bb1, g2, b2, w2, bb2, g3, b3, w3, bb3, pos);
    bias_expand_kernel<<<LTOT / 256, 256, 0, stream>>>(rpi, pos, rel);
    attn_kernel<<<512 * NUM_HEADS, 256, 0, stream>>>(qkv, rel, (float*)d_out);
}
